// GraphConvModule_51668456571363
// MI455X (gfx1250) — compile-verified, matrix operand order checked by reading, not run on hardware
//
#include <hip/hip_runtime.h>
#include <math.h>

// ---------------------------------------------------------------- types ----
typedef __bf16 bf16_t;
typedef bf16_t v16bf __attribute__((ext_vector_type(16)));
typedef float  v8f   __attribute__((ext_vector_type(8)));

// ------------------------------------------------------------ constants ----
#define TILE_M 128
#define TILE_N 128
#define TILE_K 32
#define LDS_STRIDE 40   // bf16 elems per LDS row (32 + pad vs bank conflicts)

// =================================================================== GEMM ==
// C[M,N] f32 = A[M,K](bf16,row-major) @ B[N,K](bf16,row-major)^T + bias[N]
// 256 threads = 8 waves; wave grid 4(M) x 2(N); each wave computes 32x64:
// 2 A-frags x 4 B-frags -> 8 x v_wmma_f32_16x16x32_bf16 per K step against
// 12 ds_load_b128 (vs 3:1 in the 16x32 version).
__global__ __launch_bounds__(256) void gemm_bf16_wmma(
    float* __restrict__ C, const bf16_t* __restrict__ A,
    const bf16_t* __restrict__ B, const float* __restrict__ bias,
    int M, int N, int K)
{
    __shared__ __align__(16) bf16_t sA[TILE_M * LDS_STRIDE];
    __shared__ __align__(16) bf16_t sB[TILE_N * LDS_STRIDE];

    const int tid  = threadIdx.x;
    const int lane = tid & 31;
    const int wave = tid >> 5;
    const int wm   = wave & 3;     // 0..3 : 32-row M sub-tile
    const int wn   = wave >> 2;    // 0..1 : 64-col N sub-tile
    const int lm   = lane & 15;
    const int hf   = lane >> 4;    // lane half

    const int mbase = blockIdx.y * TILE_M;
    const int nbase = blockIdx.x * TILE_N;

    v8f acc[2][4];
    for (int i = 0; i < 2; ++i)
        for (int j = 0; j < 4; ++j) acc[i][j] = (v8f){};

    // cooperative global->LDS: 128 rows x 64B per matrix per K step;
    // each thread moves two 16B chunks per matrix.
    const int ldr = tid >> 1;              // tile row 0..127
    const int ldc = (tid & 1) * 16;        // bf16 col 0 or 16
    int arow = mbase + ldr; if (arow >= M) arow = M - 1;
    int brow = nbase + ldr; if (brow >= N) brow = N - 1;
    const bf16_t* aptr = A + (size_t)arow * K + ldc;
    const bf16_t* bptr = B + (size_t)brow * K + ldc;
    bf16_t* sAw = &sA[ldr * LDS_STRIDE + ldc];
    bf16_t* sBw = &sB[ldr * LDS_STRIDE + ldc];

    for (int kt = 0; kt < K; kt += TILE_K) {
        __syncthreads();
        *(uint4*)(sAw)     = *(const uint4*)(aptr + kt);
        *(uint4*)(sAw + 8) = *(const uint4*)(aptr + kt + 8);
        *(uint4*)(sBw)     = *(const uint4*)(bptr + kt);
        *(uint4*)(sBw + 8) = *(const uint4*)(bptr + kt + 8);
        if (kt + TILE_K < K) {                     // CDNA5 global_prefetch_b8
            __builtin_prefetch(aptr + kt + TILE_K, 0, 1);
            __builtin_prefetch(bptr + kt + TILE_K, 0, 1);
        }
        __syncthreads();

        // Fragments per CDNA5 16-bit 16x32 layout:
        // lane half 0: K = {0..7, 16..23};  half 1: K = {8..15, 24..31}
        union U { uint4 q[2]; v16bf v; };
        U fa[2], fb[4];
        for (int i = 0; i < 2; ++i) {
            const bf16_t* p = &sA[(wm * 32 + i * 16 + lm) * LDS_STRIDE];
            fa[i].q[0] = *(const uint4*)(p + hf * 8);
            fa[i].q[1] = *(const uint4*)(p + 16 + hf * 8);
        }
        for (int j = 0; j < 4; ++j) {
            const bf16_t* p = &sB[(wn * 64 + j * 16 + lm) * LDS_STRIDE];
            fb[j].q[0] = *(const uint4*)(p + hf * 8);
            fb[j].q[1] = *(const uint4*)(p + 16 + hf * 8);
        }
        for (int i = 0; i < 2; ++i)
            for (int j = 0; j < 4; ++j)
                acc[i][j] = __builtin_amdgcn_wmma_f32_16x16x32_bf16(
                    false, fa[i].v, false, fb[j].v, (short)0, acc[i][j],
                    false, false);
    }

    // C/D layout: lanes 0-15 -> M=r, lanes 16-31 -> M=r+8; N = lane&15
    for (int i = 0; i < 2; ++i) {
        const int rowb = mbase + wm * 32 + i * 16 + hf * 8;
        for (int j = 0; j < 4; ++j) {
            const int col = nbase + wn * 64 + j * 16 + lm;
            const float bz = bias[col];
            union { v8f v; float f[8]; } u; u.v = acc[i][j];
            for (int r = 0; r < 8; ++r) {
                const int row = rowb + r;
                if (row < M) C[(size_t)row * N + col] = u.f[r] + bz;
            }
        }
    }
}

// ====================================================== conversions etc. ===
__global__ void f32_to_bf16_pad(const float* __restrict__ src,
                                bf16_t* __restrict__ dst,
                                long rows, int K, int Kp)
{
    const long total = rows * (long)Kp;
    for (long i = (long)blockIdx.x * blockDim.x + threadIdx.x; i < total;
         i += (long)gridDim.x * blockDim.x) {
        if (K == Kp) { dst[i] = (bf16_t)src[i]; }
        else {
            const long r = i / Kp; const int c = (int)(i - r * Kp);
            dst[i] = (c < K) ? (bf16_t)src[r * (long)K + c] : (bf16_t)0.0f;
        }
    }
}

__global__ void gather_emb_bf16(const float* __restrict__ emb,
                                const int* __restrict__ nid,
                                bf16_t* __restrict__ dst, long n)
{
    const long total = n * 256;
    for (long i = (long)blockIdx.x * blockDim.x + threadIdx.x; i < total;
         i += (long)gridDim.x * blockDim.x) {
        const long r = i >> 8; const int c = (int)(i & 255);
        dst[i] = (bf16_t)emb[((size_t)(nid[r] + 1)) * 256 + c];
    }
}

__global__ void gelu_to_bf16(const float* __restrict__ src,
                             bf16_t* __restrict__ dst, long n)
{
    for (long i = (long)blockIdx.x * blockDim.x + threadIdx.x; i < n;
         i += (long)gridDim.x * blockDim.x) {
        const float x = src[i];
        dst[i] = (bf16_t)(0.5f * x * (1.0f + erff(x * 0.70710678118654752f)));
    }
}

// mem f32 (n2 x 1024) -> bf16 (2*n2 x 512): slot s = col>>9 goes to row s*n2+n
__global__ void mem_to_bf16(const float* __restrict__ src,
                            bf16_t* __restrict__ dst, int n2)
{
    const long total = (long)n2 * 1024;
    for (long i = (long)blockIdx.x * blockDim.x + threadIdx.x; i < total;
         i += (long)gridDim.x * blockDim.x) {
        const int n = (int)(i >> 10); const int c = (int)(i & 1023);
        const int slot = c >> 9;      const int col = c & 511;
        dst[((size_t)(slot * n2 + n)) * 512 + col] = (bf16_t)src[i];
    }
}

// =============================================== row-wise (wave per row) ===
// h = nh + LayerNorm(leaky_relu(cp,0.1))*g+b ; h /= max(||h||,1e-5)
__global__ __launch_bounds__(256) void node_h_epilogue(
    const float* __restrict__ nh, const float* __restrict__ cp,
    const float* __restrict__ g,  const float* __restrict__ beta,
    float* __restrict__ out, int rows, int ldo, int colOff)
{
    const int row = blockIdx.x * 8 + (threadIdx.x >> 5);
    if (row >= rows) return;
    const int lane = threadIdx.x & 31;
    const float* cpr = cp + (size_t)row * 512;
    const float* nhr = nh + (size_t)row * 512;
    float c[16]; float s = 0.f, ss = 0.f;
    for (int e = 0; e < 16; ++e) {
        float x = cpr[lane + 32 * e];
        x = (x > 0.f) ? x : 0.1f * x;
        c[e] = x; s += x; ss += x * x;
    }
    for (int m = 16; m >= 1; m >>= 1) {
        s += __shfl_xor(s, m, 32); ss += __shfl_xor(ss, m, 32);
    }
    const float mean = s * (1.0f / 512.0f);
    const float var  = ss * (1.0f / 512.0f) - mean * mean;
    const float rinv = rsqrtf(var + 1e-5f);
    float h[16]; float n2s = 0.f;
    for (int e = 0; e < 16; ++e) {
        const int col = lane + 32 * e;
        const float v = (c[e] - mean) * rinv * g[col] + beta[col] + nhr[col];
        h[e] = v; n2s += v * v;
    }
    for (int m = 16; m >= 1; m >>= 1) n2s += __shfl_xor(n2s, m, 32);
    const float inv = 1.0f / fmaxf(sqrtf(n2s), 1e-5f);
    float* o = out + (size_t)row * ldo + colOff;
    for (int e = 0; e < 16; ++e) o[lane + 32 * e] = h[e] * inv;
}

__global__ __launch_bounds__(256) void agg_mean10(
    const float* __restrict__ src, const int* __restrict__ idx,
    float* __restrict__ dst, int rows, int cols, int ldsrc, int lddst)
{
    const int row = blockIdx.x * 8 + (threadIdx.x >> 5);
    if (row >= rows) return;
    const int lane = threadIdx.x & 31;
    int id[10];
    for (int j = 0; j < 10; ++j) id[j] = idx[(size_t)row * 10 + j];
    for (int c = lane; c < cols; c += 32) {
        float a = 0.f;
        for (int j = 0; j < 10; ++j) a += src[(size_t)id[j] * ldsrc + c];
        dst[(size_t)row * lddst + c] = a * 0.1f;
    }
}

__global__ __launch_bounds__(256) void row_normalize(
    float* __restrict__ x, int rows, int cols)
{
    const int row = blockIdx.x * 8 + (threadIdx.x >> 5);
    if (row >= rows) return;
    const int lane = threadIdx.x & 31;
    float* xr = x + (size_t)row * cols;
    float ss = 0.f;
    for (int c = lane; c < cols; c += 32) { const float v = xr[c]; ss += v * v; }
    for (int m = 16; m >= 1; m >>= 1) ss += __shfl_xor(ss, m, 32);
    const float inv = 1.0f / fmaxf(sqrtf(ss), 1e-5f);
    for (int c = lane; c < cols; c += 32) xr[c] *= inv;
}

__global__ __launch_bounds__(256) void row_normalize_out(
    const float* __restrict__ x, float* __restrict__ y, int rows, int cols)
{
    const int row = blockIdx.x * 8 + (threadIdx.x >> 5);
    if (row >= rows) return;
    const int lane = threadIdx.x & 31;
    const float* xr = x + (size_t)row * cols;
    float* yr = y + (size_t)row * cols;
    float ss = 0.f;
    for (int c = lane; c < cols; c += 32) { const float v = xr[c]; ss += v * v; }
    for (int m = 16; m >= 1; m >>= 1) ss += __shfl_xor(ss, m, 32);
    const float inv = 1.0f / fmaxf(sqrtf(ss), 1e-5f);
    for (int c = lane; c < cols; c += 32) yr[c] = xr[c] * inv;
}

// x = LayerNorm(x + t)*g + b   (rows x 512)
__global__ __launch_bounds__(256) void add_ln_512(
    float* __restrict__ x, const float* __restrict__ t,
    const float* __restrict__ g, const float* __restrict__ b, int rows)
{
    const int row = blockIdx.x * 8 + (threadIdx.x >> 5);
    if (row >= rows) return;
    const int lane = threadIdx.x & 31;
    float* xr = x + (size_t)row * 512;
    const float* tr = t + (size_t)row * 512;
    float vb[16]; float s = 0.f, ss = 0.f;
    for (int e = 0; e < 16; ++e) {
        const int c = lane + 32 * e;
        const float v = xr[c] + tr[c];
        vb[e] = v; s += v; ss += v * v;
    }
    for (int m = 16; m >= 1; m >>= 1) {
        s += __shfl_xor(s, m, 32); ss += __shfl_xor(ss, m, 32);
    }
    const float mean = s * (1.0f / 512.0f);
    const float var  = ss * (1.0f / 512.0f) - mean * mean;
    const float r    = rsqrtf(var + 1e-5f);
    for (int e = 0; e < 16; ++e) {
        const int c = lane + 32 * e;
        xr[c] = (vb[e] - mean) * r * g[c] + b[c];
    }
}

// Cross-attention: 2 keys, NHEAD=2, dh=256. wave per query row; lanes 0-15
// handle head 0, 16-31 head 1; shuffle-reduce dot products within each half.
__global__ __launch_bounds__(256) void ca_combine(
    const float* __restrict__ q, const float* __restrict__ k,
    const float* __restrict__ v, float* __restrict__ o, int n2)
{
    const int row = blockIdx.x * 8 + (threadIdx.x >> 5);
    if (row >= n2) return;
    const int lane = threadIdx.x & 31;
    const int head = lane >> 4;
    const int l16  = lane & 15;
    const float* qr = q + (size_t)row * 512 + head * 256;
    const float* k0 = k + (size_t)row * 512 + head * 256;
    const float* k1 = k + (size_t)(row + n2) * 512 + head * 256;
    const float* v0 = v + (size_t)row * 512 + head * 256;
    const float* v1 = v + (size_t)(row + n2) * 512 + head * 256;
    float d0 = 0.f, d1 = 0.f;
    for (int e = 0; e < 16; ++e) {
        const int c = l16 + e * 16;
        const float qq = qr[c];
        d0 += qq * k0[c]; d1 += qq * k1[c];
    }
    for (int m = 8; m >= 1; m >>= 1) {
        d0 += __shfl_xor(d0, m, 16); d1 += __shfl_xor(d1, m, 16);
    }
    const float s0 = d0 * 0.0625f, s1 = d1 * 0.0625f;  // dh^-0.5 = 1/16
    const float mx = fmaxf(s0, s1);
    const float e0 = expf(s0 - mx), e1 = expf(s1 - mx);
    const float a0 = e0 / (e0 + e1), a1 = e1 / (e0 + e1);
    float* orow = o + (size_t)row * 512 + head * 256;
    for (int e = 0; e < 16; ++e) {
        const int c = l16 + e * 16;
        orow[c] = a0 * v0[c] + a1 * v1[c];
    }
}

// ================================================================= host ====
static inline int gs_blocks(long total) {
    long b = (total + 255) / 256;
    if (b > 65535) b = 65535;
    if (b < 1) b = 1;
    return (int)b;
}

extern "C" void kernel_launch(void* const* d_in, const int* in_sizes, int n_in,
                              void* d_out, int out_size, void* d_ws, size_t ws_size,
                              hipStream_t stream)
{
    (void)n_in; (void)out_size; (void)ws_size;
    // ---- inputs (setup_inputs order) ----
    const int*   nid0     = (const int*)  d_in[0];
    const int*   nid1     = (const int*)  d_in[1];
    const int*   nid2     = (const int*)  d_in[2];
    const int*   esrc0    = (const int*)  d_in[3];
    const int*   esrc1    = (const int*)  d_in[5];
    const float* content0 = (const float*)d_in[7];
    const float* content1 = (const float*)d_in[8];
    const float* content2 = (const float*)d_in[9];
    const float* node_emb = (const float*)d_in[10];
    const float* ee_w     = (const float*)d_in[11];
    const float* ee_b     = (const float*)d_in[12];
    const float* proj_w   = (const float*)d_in[13];
    const float* proj_b   = (const float*)d_in[14];
    const float* proj_g   = (const float*)d_in[15];
    const float* proj_bt  = (const float*)d_in[16];
    const float* sa_in_w  = (const float*)d_in[17];
    const float* sa_in_b  = (const float*)d_in[18];
    const float* sa_out_w = (const float*)d_in[19];
    const float* sa_out_b = (const float*)d_in[20];
    const float* ca_in_w  = (const float*)d_in[21];
    const float* ca_in_b  = (const float*)d_in[22];
    const float* ca_out_w = (const float*)d_in[23];
    const float* ca_out_b = (const float*)d_in[24];
    const float* lin1_w   = (const float*)d_in[25];
    const float* lin1_b   = (const float*)d_in[26];
    const float* lin2_w   = (const float*)d_in[27];
    const float* lin2_b   = (const float*)d_in[28];
    const float* n1_g = (const float*)d_in[29];
    const float* n1_b = (const float*)d_in[30];
    const float* n2_g = (const float*)d_in[31];
    const float* n2_b = (const float*)d_in[32];
    const float* n3_g = (const float*)d_in[33];
    const float* n3_b = (const float*)d_in[34];

    const int N0 = in_sizes[0], N1 = in_sizes[1], N2 = in_sizes[2];

    // ---- workspace carve ----
    char* wp = (char*)d_ws;
    auto carve = [&](size_t bytes) -> void* {
        uintptr_t a = (uintptr_t)wp;
        a = (a + 255) & ~(uintptr_t)255;
        wp = (char*)(a + bytes);
        return (void*)a;
    };
    bf16_t* wEE    = (bf16_t*)carve((size_t)512 * 256 * 2);
    bf16_t* wPROJ  = (bf16_t*)carve((size_t)512 * 320 * 2);
    bf16_t* wSAin  = (bf16_t*)carve((size_t)2 * 1536 * 512 * 2);
    bf16_t* wSAout = (bf16_t*)carve((size_t)2 * 512 * 512 * 2);
    bf16_t* wCAin  = (bf16_t*)carve((size_t)2 * 1536 * 512 * 2);
    bf16_t* wCAout = (bf16_t*)carve((size_t)2 * 512 * 512 * 2);
    bf16_t* wL1    = (bf16_t*)carve((size_t)2 * 2048 * 512 * 2);
    bf16_t* wL2    = (bf16_t*)carve((size_t)2 * 512 * 2048 * 2);
    bf16_t* embg   = (bf16_t*)carve((size_t)N0 * 256 * 2);
    bf16_t* contb  = (bf16_t*)carve((size_t)N0 * 320 * 2);
    float*  t_nh   = (float*) carve((size_t)N0 * 512 * 4);
    float*  t_cp   = (float*) carve((size_t)N0 * 512 * 4);
    float*  h0     = (float*) carve((size_t)N0 * 512 * 4);
    float*  h1cat  = (float*) carve((size_t)N1 * 1024 * 4);
    float*  tgt    = (float*) carve((size_t)N2 * 512 * 4);
    float*  mem_f  = (float*) carve((size_t)N2 * 1024 * 4);
    float*  t2f    = (float*) carve((size_t)N2 * 512 * 4);
    float*  qf     = (float*) carve((size_t)N2 * 512 * 4);
    float*  kf     = (float*) carve((size_t)2 * N2 * 512 * 4);
    float*  vf     = (float*) carve((size_t)2 * N2 * 512 * 4);
    float*  of     = (float*) carve((size_t)N2 * 512 * 4);
    float*  fff    = (float*) carve((size_t)N2 * 2048 * 4);
    bf16_t* xbf    = (bf16_t*)carve((size_t)N2 * 2048 * 2);
    bf16_t* membf  = (bf16_t*)carve((size_t)2 * N2 * 512 * 2);

    auto conv = [&](const float* s, bf16_t* d, long rows, int K, int Kp) {
        f32_to_bf16_pad<<<gs_blocks(rows * (long)Kp), 256, 0, stream>>>(s, d, rows, K, Kp);
    };
    auto gemm = [&](float* C, const bf16_t* A, const bf16_t* B,
                    const float* bias, int M, int N, int K) {
        dim3 g(N / TILE_N, (M + TILE_M - 1) / TILE_M);
        gemm_bf16_wmma<<<g, 256, 0, stream>>>(C, A, B, bias, M, N, K);
    };

    // ---- weight conversions (f32 -> bf16, pad proj 300->320) ----
    conv(ee_w,     wEE,    512, 256, 256);
    conv(proj_w,   wPROJ,  512, 300, 320);
    conv(sa_in_w,  wSAin,  2 * 1536, 512, 512);
    conv(sa_out_w, wSAout, 2 * 512,  512, 512);
    conv(ca_in_w,  wCAin,  2 * 1536, 512, 512);
    conv(ca_out_w, wCAout, 2 * 512,  512, 512);
    conv(lin1_w,   wL1,    2 * 2048, 512, 512);
    conv(lin2_w,   wL2,    2 * 512,  2048, 2048);

    // ---- node_h for the three node sets ----
    auto node_set = [&](const int* nid, int Ns, const float* content,
                        float* outp, int ldo, int colOff) {
        gather_emb_bf16<<<gs_blocks((long)Ns * 256), 256, 0, stream>>>(
            node_emb, nid, embg, Ns);
        conv(content, contb, Ns, 300, 320);
        gemm(t_nh, embg,  wEE,   ee_b,   Ns, 512, 256);
        gemm(t_cp, contb, wPROJ, proj_b, Ns, 512, 320);
        node_h_epilogue<<<(Ns + 7) / 8, 256, 0, stream>>>(
            t_nh, t_cp, proj_g, proj_bt, outp, Ns, ldo, colOff);
    };
    node_set(nid0, N0, content0, h0,    512,  0);
    node_set(nid1, N1, content1, h1cat, 1024, 512);  // h1 -> right half of concat
    node_set(nid2, N2, content2, tgt,   512,  0);    // tgt = h2

    // ---- graph aggregation ----
    agg_mean10<<<(N1 + 7) / 8, 256, 0, stream>>>(h0, esrc0, h1cat, N1, 512, 512, 1024);
    row_normalize<<<(N1 + 7) / 8, 256, 0, stream>>>(h1cat, N1, 1024);
    agg_mean10<<<(N2 + 7) / 8, 256, 0, stream>>>(h1cat, esrc1, mem_f, N2, 1024, 1024, 1024);
    mem_to_bf16<<<gs_blocks((long)N2 * 1024), 256, 0, stream>>>(mem_f, membf, N2);

    // ---- 2-layer transformer decoder (seq-len-1 SA degenerates to V path) --
    for (int l = 0; l < 2; ++l) {
        const size_t w3 = (size_t)l * 1536 * 512;
        const size_t w1 = (size_t)l * 512 * 512;
        // self-attn: softmax over 1 key == 1  ->  (tgt@Wv+bv)@Wout+b
        conv(tgt, xbf, N2, 512, 512);
        gemm(t2f, xbf, wSAin + w3 + (size_t)1024 * 512,
             sa_in_b + (size_t)l * 1536 + 1024, N2, 512, 512);
        conv(t2f, xbf, N2, 512, 512);
        gemm(of, xbf, wSAout + w1, sa_out_b + (size_t)l * 512, N2, 512, 512);
        add_ln_512<<<(N2 + 7) / 8, 256, 0, stream>>>(
            tgt, of, n1_g + (size_t)l * 512, n1_b + (size_t)l * 512, N2);
        // cross-attn: 2 memory slots per query
        conv(tgt, xbf, N2, 512, 512);
        gemm(qf, xbf,   wCAin + w3,                      ca_in_b + (size_t)l * 1536,        N2,     512, 512);
        gemm(kf, membf, wCAin + w3 + (size_t)512 * 512,  ca_in_b + (size_t)l * 1536 + 512,  2 * N2, 512, 512);
        gemm(vf, membf, wCAin + w3 + (size_t)1024 * 512, ca_in_b + (size_t)l * 1536 + 1024, 2 * N2, 512, 512);
        ca_combine<<<(N2 + 7) / 8, 256, 0, stream>>>(qf, kf, vf, of, N2);
        conv(of, xbf, N2, 512, 512);
        gemm(t2f, xbf, wCAout + w1, ca_out_b + (size_t)l * 512, N2, 512, 512);
        add_ln_512<<<(N2 + 7) / 8, 256, 0, stream>>>(
            tgt, t2f, n2_g + (size_t)l * 512, n2_b + (size_t)l * 512, N2);
        // feed-forward with exact gelu
        conv(tgt, xbf, N2, 512, 512);
        gemm(fff, xbf, wL1 + (size_t)l * 2048 * 512,
             lin1_b + (size_t)l * 2048, N2, 2048, 512);
        gelu_to_bf16<<<gs_blocks((long)N2 * 2048), 256, 0, stream>>>(
            fff, xbf, (long)N2 * 2048);
        gemm(t2f, xbf, wL2 + (size_t)l * 512 * 2048,
             lin2_b + (size_t)l * 512, N2, 512, 2048);
        add_ln_512<<<(N2 + 7) / 8, 256, 0, stream>>>(
            tgt, t2f, n3_g + (size_t)l * 512, n3_b + (size_t)l * 512, N2);
    }

    // ---- final row L2-normalize -> output ----
    row_normalize_out<<<(N2 + 7) / 8, 256, 0, stream>>>(tgt, (float*)d_out, N2, 512);
}
